// SelfAttention_45938970198708
// MI455X (gfx1250) — matrix operand order repaired and checked by reading, not compiled, not run on hardware
//
#include <hip/hip_runtime.h>
#include <hip/hip_bf16.h>

typedef _Float16 half_t;
typedef __attribute__((ext_vector_type(16))) _Float16 v16h;
typedef __attribute__((ext_vector_type(8)))  _Float16 v8h;
typedef __attribute__((ext_vector_type(8)))  float    v8f;

#define B_  8
#define T_  1024
#define C_  768
#define H_  12
#define HD_ 64
#define M_  (B_ * T_)   // 8192 rows for the projection GEMMs

// ---------------------------------------------------------------------------
// WMMA fragment helpers (CDNA5 wave32 layouts, ISA 7.12.2)
// A-matrix 16x32 f16: lane l holds row m=l&15; halves idx0..7 -> K=g*8..g*8+7,
// idx8..15 -> K=16+g*8..16+g*8+7 (g = l>>4). Two 16B contiguous loads.
// B-matrix 32x16 f16 loaded from transposed storage BT[n][k]: lane l holds
// column n=l&15, same K order as A (idx0..7 -> K=g*8.., idx8..15 -> K=16+g*8..).
// ---------------------------------------------------------------------------
__device__ __forceinline__ v16h load_a_frag(const half_t* p, int ld) {
  const int lane = threadIdx.x & 31;
  const int row  = lane & 15;
  const int g    = lane >> 4;
  const v8h lo = *(const v8h*)(p + (size_t)row * ld + g * 8);
  const v8h hi = *(const v8h*)(p + (size_t)row * ld + 16 + g * 8);
  v16h a;
#pragma unroll
  for (int i = 0; i < 8; ++i) { a[i] = lo[i]; a[8 + i] = hi[i]; }
  return a;
}

__device__ __forceinline__ v16h load_bt_frag(const half_t* p, int ld) {
  const int lane = threadIdx.x & 31;
  const int n    = lane & 15;
  const int g    = lane >> 4;
  const half_t* r = p + (size_t)n * ld;
  const v8h lo = *(const v8h*)(r + g * 8);
  const v8h hi = *(const v8h*)(r + 16 + g * 8);
  v16h b;
#pragma unroll
  for (int i = 0; i < 8; ++i) { b[i] = lo[i]; b[8 + i] = hi[i]; }
  return b;
}

__device__ __forceinline__ v8f wmma16x16x32(v16h a, v16h b, v8f c) {
  return __builtin_amdgcn_wmma_f32_16x16x32_f16(
      /*neg_a=*/false, a, /*neg_b=*/false, b,
      /*c_mod=*/(short)0, c, /*reuse_a=*/false, /*reuse_b=*/false);
}

// Async global -> LDS copy of 16 bytes (tracked by ASYNCcnt).
// vdst = 32-bit LDS byte address (generic LDS ptr truncates to LDS offset,
// ISA 10.2 aperture rules), vaddr = 64-bit global address.
__device__ __forceinline__ void async_copy_b128(void* lds, const void* gptr) {
  const unsigned lds_off = (unsigned)(uintptr_t)lds;
  const unsigned long long ga = (unsigned long long)(uintptr_t)gptr;
  asm volatile("global_load_async_to_lds_b128 %0, %1, off"
               :: "v"(lds_off), "v"(ga) : "memory");
}

// ---------------------------------------------------------------------------
// Precision conversion / weight transpose
// ---------------------------------------------------------------------------
__global__ void __launch_bounds__(256)
f32_to_f16_kernel(const float* __restrict__ src, half_t* __restrict__ dst, int n) {
  int i = blockIdx.x * 256 + threadIdx.x;
  if (i < n) dst[i] = (half_t)src[i];
}

// dst[n*C + k] = (f16) src[k*C + n]   (weight -> transposed f16)
__global__ void __launch_bounds__(256)
transpose_w_kernel(const float* __restrict__ src, half_t* __restrict__ dst) {
  int i = blockIdx.x * 256 + threadIdx.x;
  if (i < C_ * C_) {
    int nn = i / C_, kk = i % C_;
    dst[i] = (half_t)src[(size_t)kk * C_ + nn];
  }
}

// ---------------------------------------------------------------------------
// Fused QKV projection: q/k/v = x @ w + b.
// Block = 4 waves (2x2) on a 64x128 tile; per 32-wide K-step the block stages
// A (64x32) and BT (128x32) into double-buffered LDS via async copies that
// overlap the previous step's WMMAs. Each wave computes 32x64 (8 WMMAs/step)
// from LDS fragments. Q,K stored [B,H,T,HD]; V stored transposed [B,H,HD,T].
// ---------------------------------------------------------------------------
__global__ void __launch_bounds__(128)
qkv_gemm_kernel(const half_t* __restrict__ xh,
                const half_t* __restrict__ wqT, const half_t* __restrict__ wkT,
                const half_t* __restrict__ wvT,
                const float* __restrict__ bq, const float* __restrict__ bk,
                const float* __restrict__ bv,
                half_t* __restrict__ Q, half_t* __restrict__ K,
                half_t* __restrict__ Vt) {
  __shared__ __align__(16) half_t sA[2][64 * 32];    // [buf][row][k]
  __shared__ __align__(16) half_t sB[2][128 * 32];   // [buf][col][k]

  const int tid   = threadIdx.x;
  const int lane  = tid & 31;
  const int wave  = tid >> 5;
  const int waveM = wave >> 1;                 // 0..1
  const int waveN = wave & 1;                  // 0..1

  const int row0 = blockIdx.x * 64;            // M base (0..8128)
  const int colg = blockIdx.y * 128;           // col base across q|k|v (0..2176)
  const int w    = colg / C_;                  // 0=q,1=k,2=v (128 | 768, no straddle)
  const int col0 = colg - w * C_;              // col base within this weight

  const half_t* wT   = (w == 0) ? wqT : (w == 1) ? wkT : wvT;
  const float*  bias = (w == 0) ? bq  : (w == 1) ? bk  : bv;

  const half_t* pa = xh + (size_t)row0 * C_;
  const half_t* pb = wT + (size_t)col0 * C_;

  // stage one K-step into LDS buffer `buf` (6 async b128 ops per thread)
  auto stage = [&](int kk, int buf) {
#pragma unroll
    for (int i = 0; i < 2; ++i) {              // A: 64 rows x 4 chunks
      const int id = tid + 128 * i;
      const int r = id >> 2, part = id & 3;
      async_copy_b128(&sA[buf][r * 32 + part * 8],
                      pa + (size_t)r * C_ + kk + part * 8);
    }
#pragma unroll
    for (int i = 0; i < 4; ++i) {              // BT: 128 cols x 4 chunks
      const int id = tid + 128 * i;
      const int r = id >> 2, part = id & 3;
      async_copy_b128(&sB[buf][r * 32 + part * 8],
                      pb + (size_t)r * C_ + kk + part * 8);
    }
  };

  v8f acc[8] = {};
  stage(0, 0);

  for (int kt = 0; kt < C_ / 32; ++kt) {
    asm volatile("s_wait_asynccnt 0x0" ::: "memory");  // own copies landed
    __syncthreads();                                   // everyone's landed
    if (kt + 1 < C_ / 32) stage((kt + 1) * 32, (kt + 1) & 1);

    const half_t* tA = &sA[kt & 1][0];
    const half_t* tB = &sB[kt & 1][0];
    const v16h a0 = load_a_frag(tA + (waveM * 32 +  0) * 32, 32);
    const v16h a1 = load_a_frag(tA + (waveM * 32 + 16) * 32, 32);
    const v16h b0 = load_bt_frag(tB + (waveN * 64 +  0) * 32, 32);
    const v16h b1 = load_bt_frag(tB + (waveN * 64 + 16) * 32, 32);
    const v16h b2 = load_bt_frag(tB + (waveN * 64 + 32) * 32, 32);
    const v16h b3 = load_bt_frag(tB + (waveN * 64 + 48) * 32, 32);
    acc[0] = wmma16x16x32(a0, b0, acc[0]);
    acc[1] = wmma16x16x32(a0, b1, acc[1]);
    acc[2] = wmma16x16x32(a0, b2, acc[2]);
    acc[3] = wmma16x16x32(a0, b3, acc[3]);
    acc[4] = wmma16x16x32(a1, b0, acc[4]);
    acc[5] = wmma16x16x32(a1, b1, acc[5]);
    acc[6] = wmma16x16x32(a1, b2, acc[6]);
    acc[7] = wmma16x16x32(a1, b3, acc[7]);
  }

  const int g = lane >> 4;
  const int n = lane & 15;
  const int rowbase = row0 + waveM * 32;
#pragma unroll
  for (int mh = 0; mh < 2; ++mh) {
#pragma unroll
    for (int s = 0; s < 4; ++s) {
      const int   c  = col0 + waveN * 64 + s * 16 + n;  // col within weight
      const float bc = bias[c];
      const int   h  = c >> 6;            // c / HD
      const int   d  = c & 63;            // c % HD
#pragma unroll
      for (int r = 0; r < 8; ++r) {
        const int    mrow = rowbase + mh * 16 + r + 8 * g;
        const int    bi   = mrow >> 10;   // / T
        const int    t    = mrow & 1023;  // % T
        const half_t val  = (half_t)(acc[mh * 4 + s][r] + bc);
        if (w == 2) {
          Vt[(((size_t)bi * H_ + h) * HD_ + d) * T_ + t] = val;
        } else if (w == 0) {
          Q[(((size_t)bi * H_ + h) * T_ + t) * HD_ + d] = val;
        } else {
          K[(((size_t)bi * H_ + h) * T_ + t) * HD_ + d] = val;
        }
      }
    }
  }
}

// ---------------------------------------------------------------------------
// Flash attention: each wave owns 16 query rows of one (b,h). Iterates 32-key
// blocks up to the causal diagonal with online softmax; P staged in LDS f16.
// K-fragments for block i+1 and V-fragments for block i are prefetched right
// after the S WMMAs so they overlap the softmax VALU work.
// ---------------------------------------------------------------------------
#define LOAD_KFRAGS(A00, A01, A10, A11, S0)                                  \
  A00 = load_bt_frag(Kp + (size_t)(S0) * HD_,            HD_);               \
  A01 = load_bt_frag(Kp + (size_t)(S0) * HD_ + 32,       HD_);               \
  A10 = load_bt_frag(Kp + (size_t)((S0) + 16) * HD_,      HD_);              \
  A11 = load_bt_frag(Kp + (size_t)((S0) + 16) * HD_ + 32, HD_);

__global__ void __launch_bounds__(128)
flash_attn_kernel(const half_t* __restrict__ Q, const half_t* __restrict__ K,
                  const half_t* __restrict__ Vt, half_t* __restrict__ att) {
  __shared__ __align__(16) half_t lds_p[4][16 * 32];

  const int lane  = threadIdx.x & 31;
  const int wave  = threadIdx.x >> 5;
  const int bh    = blockIdx.x;                 // 0..95
  const int qt    = blockIdx.y * 4 + wave;      // 0..63
  const int bi    = bh / H_;
  const int h     = bh % H_;
  const int qbase = qt * 16;

  const half_t* Qp = Q  + ((size_t)bi * H_ + h) * T_ * HD_;
  const half_t* Kp = K  + ((size_t)bi * H_ + h) * T_ * HD_;
  const half_t* Vp = Vt + ((size_t)bi * H_ + h) * HD_ * T_;

  // Q fragments loaded once, reused across all key blocks (HD=64 -> 2 K-steps)
  const v16h aq0 = load_a_frag(Qp + (size_t)qbase * HD_ +  0, HD_);
  const v16h aq1 = load_a_frag(Qp + (size_t)qbase * HD_ + 32, HD_);

  const int g = lane >> 4;
  const int n = lane & 15;

  v8f o[4] = {};
  float mrow[8], lrow[8];
#pragma unroll
  for (int r = 0; r < 8; ++r) { mrow[r] = -3.0e38f; lrow[r] = 0.0f; }

  half_t* P = &lds_p[wave][0];
  const int nblk = (qbase + 16 + 31) / 32;      // key blocks up to the diagonal

  v16h kc00, kc01, kc10, kc11;                  // current K fragments
  LOAD_KFRAGS(kc00, kc01, kc10, kc11, 0)

  for (int blk = 0; blk < nblk; ++blk) {
    const int s0 = blk * 32;

    // ---- S = (Q K^T) / sqrt(HD): two 16-col subtiles, 2 K-steps each ----
    v8f sacc[2];
    {
      v8f c0 = {};
      c0 = wmma16x16x32(aq0, kc00, c0);
      c0 = wmma16x16x32(aq1, kc01, c0);
      sacc[0] = c0;
      v8f c1 = {};
      c1 = wmma16x16x32(aq0, kc10, c1);
      c1 = wmma16x16x32(aq1, kc11, c1);
      sacc[1] = c1;
    }

    // ---- prefetch next block's K fragments + this block's V fragments ----
    v16h kn00, kn01, kn10, kn11;
    if (blk + 1 < nblk) { LOAD_KFRAGS(kn00, kn01, kn10, kn11, s0 + 32) }
    v16h vb0 = load_bt_frag(Vp + (size_t)( 0) * T_ + s0, T_);
    v16h vb1 = load_bt_frag(Vp + (size_t)(16) * T_ + s0, T_);
    v16h vb2 = load_bt_frag(Vp + (size_t)(32) * T_ + s0, T_);
    v16h vb3 = load_bt_frag(Vp + (size_t)(48) * T_ + s0, T_);

    // ---- causal mask + per-row block max ----
    float bmax[8];
#pragma unroll
    for (int r = 0; r < 8; ++r) {
      const int trow = qbase + r + 8 * g;
      float v0 = (float)sacc[0][r] * 0.125f;    // 1/sqrt(64)
      float v1 = (float)sacc[1][r] * 0.125f;
      if (s0 + n      > trow) v0 = -3.0e38f;
      if (s0 + 16 + n > trow) v1 = -3.0e38f;
      sacc[0][r] = v0; sacc[1][r] = v1;
      bmax[r] = fmaxf(v0, v1);
    }
#pragma unroll
    for (int msk = 1; msk < 16; msk <<= 1) {
#pragma unroll
      for (int r = 0; r < 8; ++r)
        bmax[r] = fmaxf(bmax[r], __shfl_xor(bmax[r], msk, 32));
    }

    // ---- online softmax update + stage P (f16) in LDS ----
#pragma unroll
    for (int r = 0; r < 8; ++r) {
      const float mnew  = fmaxf(mrow[r], bmax[r]);
      const float scale = __expf(mrow[r] - mnew);
      const float p0 = __expf(sacc[0][r] - mnew);
      const float p1 = __expf(sacc[1][r] - mnew);
      float psum = p0 + p1;
#pragma unroll
      for (int msk = 1; msk < 16; msk <<= 1) psum += __shfl_xor(psum, msk, 32);
      lrow[r] = lrow[r] * scale + psum;
      mrow[r] = mnew;
#pragma unroll
      for (int dt = 0; dt < 4; ++dt) o[dt][r] *= scale;
      P[(r + 8 * g) * 32 + n]      = (half_t)p0;
      P[(r + 8 * g) * 32 + 16 + n] = (half_t)p1;
    }
    asm volatile("s_wait_dscnt 0x0" ::: "memory");   // LDS stores -> frag loads

    // ---- O += P (16x32) x V (32x64) ----
    const v16h ap = load_a_frag(P, 32);
    o[0] = wmma16x16x32(ap, vb0, o[0]);
    o[1] = wmma16x16x32(ap, vb1, o[1]);
    o[2] = wmma16x16x32(ap, vb2, o[2]);
    o[3] = wmma16x16x32(ap, vb3, o[3]);

    if (blk + 1 < nblk) { kc00 = kn00; kc01 = kn01; kc10 = kn10; kc11 = kn11; }
  }

  // ---- normalize and emit [B,T,C] f16 (head-concatenated) ----
#pragma unroll
  for (int dt = 0; dt < 4; ++dt) {
#pragma unroll
    for (int r = 0; r < 8; ++r) {
      const int   t   = qbase + r + 8 * g;
      const float val = o[dt][r] / lrow[r];
      att[((size_t)bi * T_ + t) * C_ + h * HD_ + dt * 16 + n] = (half_t)val;
    }
  }
}

// ---------------------------------------------------------------------------
// Output projection: out = att @ wo + bo (fp32 result). Same async-LDS
// double-buffered pipeline as the QKV GEMM (64x128 block tile, 32x64/wave).
// ---------------------------------------------------------------------------
__global__ void __launch_bounds__(128)
out_gemm_kernel(const half_t* __restrict__ att, const half_t* __restrict__ woT,
                const float* __restrict__ bo, float* __restrict__ out) {
  __shared__ __align__(16) half_t sA[2][64 * 32];
  __shared__ __align__(16) half_t sB[2][128 * 32];

  const int tid   = threadIdx.x;
  const int lane  = tid & 31;
  const int wave  = tid >> 5;
  const int waveM = wave >> 1;
  const int waveN = wave & 1;

  const int row0 = blockIdx.x * 64;
  const int col0 = blockIdx.y * 128;

  const half_t* pa = att + (size_t)row0 * C_;
  const half_t* pb = woT + (size_t)col0 * C_;

  auto stage = [&](int kk, int buf) {
#pragma unroll
    for (int i = 0; i < 2; ++i) {
      const int id = tid + 128 * i;
      const int r = id >> 2, part = id & 3;
      async_copy_b128(&sA[buf][r * 32 + part * 8],
                      pa + (size_t)r * C_ + kk + part * 8);
    }
#pragma unroll
    for (int i = 0; i < 4; ++i) {
      const int id = tid + 128 * i;
      const int r = id >> 2, part = id & 3;
      async_copy_b128(&sB[buf][r * 32 + part * 8],
                      pb + (size_t)r * C_ + kk + part * 8);
    }
  };

  v8f acc[8] = {};
  stage(0, 0);

  for (int kt = 0; kt < C_ / 32; ++kt) {
    asm volatile("s_wait_asynccnt 0x0" ::: "memory");
    __syncthreads();
    if (kt + 1 < C_ / 32) stage((kt + 1) * 32, (kt + 1) & 1);

    const half_t* tA = &sA[kt & 1][0];
    const half_t* tB = &sB[kt & 1][0];
    const v16h a0 = load_a_frag(tA + (waveM * 32 +  0) * 32, 32);
    const v16h a1 = load_a_frag(tA + (waveM * 32 + 16) * 32, 32);
    const v16h b0 = load_bt_frag(tB + (waveN * 64 +  0) * 32, 32);
    const v16h b1 = load_bt_frag(tB + (waveN * 64 + 16) * 32, 32);
    const v16h b2 = load_bt_frag(tB + (waveN * 64 + 32) * 32, 32);
    const v16h b3 = load_bt_frag(tB + (waveN * 64 + 48) * 32, 32);
    acc[0] = wmma16x16x32(a0, b0, acc[0]);
    acc[1] = wmma16x16x32(a0, b1, acc[1]);
    acc[2] = wmma16x16x32(a0, b2, acc[2]);
    acc[3] = wmma16x16x32(a0, b3, acc[3]);
    acc[4] = wmma16x16x32(a1, b0, acc[4]);
    acc[5] = wmma16x16x32(a1, b1, acc[5]);
    acc[6] = wmma16x16x32(a1, b2, acc[6]);
    acc[7] = wmma16x16x32(a1, b3, acc[7]);
  }

  const int g = lane >> 4;
  const int n = lane & 15;
  const int rowbase = row0 + waveM * 32;
#pragma unroll
  for (int mh = 0; mh < 2; ++mh) {
#pragma unroll
    for (int s = 0; s < 4; ++s) {
      const int   c  = col0 + waveN * 64 + s * 16 + n;
      const float bc = bo[c];
#pragma unroll
      for (int r = 0; r < 8; ++r) {
        const int mrow = rowbase + mh * 16 + r + 8 * g;
        out[(size_t)mrow * C_ + c] = acc[mh * 4 + s][r] + bc;
      }
    }
  }
}

// ---------------------------------------------------------------------------
// Host launcher
// ---------------------------------------------------------------------------
extern "C" void kernel_launch(void* const* d_in, const int* in_sizes, int n_in,
                              void* d_out, int out_size, void* d_ws, size_t ws_size,
                              hipStream_t stream) {
  // inputs: 0=x, 1=mask(unused; causal applied analytically), 2=wq, 3=bq,
  //         4=wk, 5=bk, 6=wv, 7=bv, 8=wo, 9=bo
  const float* x  = (const float*)d_in[0];
  const float* wq = (const float*)d_in[2];
  const float* bq = (const float*)d_in[3];
  const float* wk = (const float*)d_in[4];
  const float* bk = (const float*)d_in[5];
  const float* wv = (const float*)d_in[6];
  const float* bv = (const float*)d_in[7];
  const float* wo = (const float*)d_in[8];
  const float* bo = (const float*)d_in[9];
  float* out = (float*)d_out;

  // workspace carve (each region 256B aligned); total ~68 MB
  size_t off = 0;
  auto carve = [&](size_t bytes) -> void* {
    void* p = (char*)d_ws + off;
    off += (bytes + 255) & ~(size_t)255;
    return p;
  };
  half_t* xh  = (half_t*)carve((size_t)M_ * C_ * sizeof(half_t));
  half_t* wqT = (half_t*)carve((size_t)C_ * C_ * sizeof(half_t));
  half_t* wkT = (half_t*)carve((size_t)C_ * C_ * sizeof(half_t));
  half_t* wvT = (half_t*)carve((size_t)C_ * C_ * sizeof(half_t));
  half_t* woT = (half_t*)carve((size_t)C_ * C_ * sizeof(half_t));
  half_t* Qh  = (half_t*)carve((size_t)B_ * H_ * T_ * HD_ * sizeof(half_t));
  half_t* Kh  = (half_t*)carve((size_t)B_ * H_ * T_ * HD_ * sizeof(half_t));
  half_t* Vt  = (half_t*)carve((size_t)B_ * H_ * HD_ * T_ * sizeof(half_t));
  half_t* att = (half_t*)carve((size_t)M_ * C_ * sizeof(half_t));
  (void)ws_size; (void)in_sizes; (void)n_in; (void)out_size;

  // 1) convert activations + weights to f16 (weights transposed to [n][k])
  {
    const int n = M_ * C_;
    f32_to_f16_kernel<<<dim3((n + 255) / 256), dim3(256), 0, stream>>>(x, xh, n);
    const int g = (C_ * C_ + 255) / 256;
    transpose_w_kernel<<<dim3(g), dim3(256), 0, stream>>>(wq, wqT);
    transpose_w_kernel<<<dim3(g), dim3(256), 0, stream>>>(wk, wkT);
    transpose_w_kernel<<<dim3(g), dim3(256), 0, stream>>>(wv, wvT);
    transpose_w_kernel<<<dim3(g), dim3(256), 0, stream>>>(wo, woT);
  }

  // 2) fused QKV projection: 128 x 18 blocks, 64x128 tile each
  qkv_gemm_kernel<<<dim3(M_ / 64, (3 * C_) / 128), dim3(128), 0, stream>>>(
      xh, wqT, wkT, wvT, bq, bk, bv, Qh, Kh, Vt);

  // 3) flash attention (96 (b,h) pairs x 64 query tiles)
  flash_attn_kernel<<<dim3(B_ * H_, (T_ / 16) / 4), dim3(128), 0, stream>>>(
      Qh, Kh, Vt, att);

  // 4) output projection: 128 x 6 blocks, 64x128 tile each
  out_gemm_kernel<<<dim3(M_ / 64, C_ / 128), dim3(128), 0, stream>>>(
      att, woT, bo, out);
}
